// Conv1d_81776177316274
// MI455X (gfx1250) — matrix operand order repaired and checked by reading, not compiled, not run on hardware
//
#include <hip/hip_runtime.h>
#include <hip/hip_bf16.h>

typedef __attribute__((ext_vector_type(16))) __bf16 v16bf;
typedef __attribute__((ext_vector_type(8)))  float  v8f;

#define C_IN   256
#define C_OUT  512
#define LEN    4096
#define KTAPS  9
#define PAD    4
#define BATCH  32

#define TILE_T   128                 // t-positions per block
#define TILE_CO  128                 // output channels per block
#define HALO_T   (TILE_T + 2*PAD)    // 136
#define XSTRIDE  264                 // padded LDS row stride (halfs)
#define WSTRIDE  264

union Frag {
    v16bf v;
    uint4 q[2];
};

struct Frags {
    Frag a0, a1, b0, b1, b2, b3;
};

__device__ __forceinline__ unsigned short f2bf(float f) {
    unsigned u = __float_as_uint(f);
    u += 0x7FFFu + ((u >> 16) & 1u);      // round-to-nearest-even
    return (unsigned short)(u >> 16);
}

__device__ __forceinline__ void load_frags(Frags& f,
                                           const unsigned short* Ap0,
                                           const unsigned short* Ap1,
                                           const unsigned short* Bp0,
                                           const unsigned short* Bp1,
                                           const unsigned short* Bp2,
                                           const unsigned short* Bp3,
                                           int ci0, int lhalf) {
    // A fragments: 16x32 bf16; element e -> K = 16*lhalf + e
    f.a0.q[0] = *(const uint4*)(Ap0 + ci0 + 16 * lhalf);
    f.a0.q[1] = *(const uint4*)(Ap0 + ci0 + 8 + 16 * lhalf);
    f.a1.q[0] = *(const uint4*)(Ap1 + ci0 + 16 * lhalf);
    f.a1.q[1] = *(const uint4*)(Ap1 + ci0 + 8 + 16 * lhalf);
    // B fragments: 32x16 bf16; lane = N col, element e -> K = ci0 + 16*lhalf + e
    f.b0.q[0] = *(const uint4*)(Bp0 + ci0);
    f.b0.q[1] = *(const uint4*)(Bp0 + ci0 + 8);
    f.b1.q[0] = *(const uint4*)(Bp1 + ci0);
    f.b1.q[1] = *(const uint4*)(Bp1 + ci0 + 8);
    f.b2.q[0] = *(const uint4*)(Bp2 + ci0);
    f.b2.q[1] = *(const uint4*)(Bp2 + ci0 + 8);
    f.b3.q[0] = *(const uint4*)(Bp3 + ci0);
    f.b3.q[1] = *(const uint4*)(Bp3 + ci0 + 8);
}

// Repack weight (C_OUT, C_IN, K) f32  ->  (K, C_OUT, C_IN) bf16 in workspace.
__global__ void repack_weights(const float* __restrict__ w,
                               unsigned short* __restrict__ w2) {
    int idx = blockIdx.x * 256 + threadIdx.x;
    const int total = KTAPS * C_OUT * C_IN;
    if (idx >= total) return;
    int k  = idx / (C_OUT * C_IN);
    int r  = idx - k * (C_OUT * C_IN);
    int co = r >> 8;           // / C_IN
    int ci = r & (C_IN - 1);
    w2[idx] = f2bf(w[(co * C_IN + ci) * KTAPS + k]);
}

__global__ __launch_bounds__(256)
void conv1d_wmma_bf16(const float* __restrict__ x,
                      const unsigned short* __restrict__ w2,   // (K, C_OUT, C_IN) bf16
                      const float* __restrict__ bias,
                      float* __restrict__ out) {
    __shared__ unsigned short Xs[HALO_T * XSTRIDE];   // [t_halo][ci] bf16, transposed
    __shared__ unsigned short Ws[TILE_CO * WSTRIDE];  // [co][ci] bf16, one tap at a time

    const int tid  = threadIdx.x;
    const int t0   = blockIdx.x * TILE_T;
    const int co0  = blockIdx.y * TILE_CO;
    const int b    = blockIdx.z;

    // ---- stage x slab (with halo, zero padded) into LDS, transposed + bf16 ----
    {
        const float* xb = x + (size_t)b * C_IN * LEN;
        const int nCh = HALO_T / 4;                  // 34 float4 chunks per ci row
        for (int idx = tid; idx < C_IN * nCh; idx += 256) {
            int ci = idx / nCh;
            int c  = idx - ci * nCh;
            int tg = t0 - PAD + c * 4;               // 16B aligned when in range
            float4 v;
            if (tg >= 0 && tg + 3 < LEN) {
                v = *(const float4*)&xb[ci * LEN + tg];
            } else {
                v.x = (tg + 0 >= 0 && tg + 0 < LEN) ? xb[ci * LEN + tg + 0] : 0.0f;
                v.y = (tg + 1 >= 0 && tg + 1 < LEN) ? xb[ci * LEN + tg + 1] : 0.0f;
                v.z = (tg + 2 >= 0 && tg + 2 < LEN) ? xb[ci * LEN + tg + 2] : 0.0f;
                v.w = (tg + 3 >= 0 && tg + 3 < LEN) ? xb[ci * LEN + tg + 3] : 0.0f;
            }
            int th = c * 4;
            Xs[(th + 0) * XSTRIDE + ci] = f2bf(v.x);
            Xs[(th + 1) * XSTRIDE + ci] = f2bf(v.y);
            Xs[(th + 2) * XSTRIDE + ci] = f2bf(v.z);
            Xs[(th + 3) * XSTRIDE + ci] = f2bf(v.w);
        }
    }

    // ---- wave / lane decomposition: 8 waves = 4 (co) x 2 (t) ----
    const int wave  = tid >> 5;
    const int lane  = tid & 31;
    const int wco   = wave & 3;          // co subtile: wco*32
    const int wt    = wave >> 2;         // t  subtile: wt*64 (four 16-wide tiles)
    const int lrow  = lane & 15;
    const int lhalf = lane >> 4;

    v8f acc[2][4];
#pragma unroll
    for (int i = 0; i < 2; ++i)
#pragma unroll
        for (int j = 0; j < 4; ++j) acc[i][j] = (v8f){};

    const unsigned short* Ap0 = &Ws[(wco * 32 + lrow) * WSTRIDE];
    const unsigned short* Ap1 = Ap0 + 16 * WSTRIDE;
    const unsigned WsLdsBase = (unsigned)(size_t)(void*)&Ws[0];

    for (int k = 0; k < KTAPS; ++k) {
        __syncthreads();   // protect Ws reuse from previous tap
        // ---- async-copy weight tile (128 co x 256 ci bf16, contiguous) to LDS ----
        {
            const unsigned short* wsrc = w2 + ((size_t)k * C_OUT + co0) * C_IN;
            // 4096 chunks of 16B; 16 per thread
            for (int idx = tid; idx < (TILE_CO * C_IN) / 8; idx += 256) {
                int row = idx >> 5;                  // / (C_IN/8)
                int c16 = idx & 31;
                unsigned lds = WsLdsBase + (unsigned)(row * (WSTRIDE * 2) + c16 * 16);
                const unsigned short* g = wsrc + idx * 8;
                asm volatile("global_load_async_to_lds_b128 %0, %1, off"
                             :: "v"(lds), "v"(g) : "memory");
            }
            asm volatile("s_wait_asynccnt 0x0" ::: "memory");
        }
        __syncthreads();

        const unsigned short* Bp0 = &Xs[(wt * 64 +  0 + k + lrow) * XSTRIDE + 16 * lhalf];
        const unsigned short* Bp1 = &Xs[(wt * 64 + 16 + k + lrow) * XSTRIDE + 16 * lhalf];
        const unsigned short* Bp2 = &Xs[(wt * 64 + 32 + k + lrow) * XSTRIDE + 16 * lhalf];
        const unsigned short* Bp3 = &Xs[(wt * 64 + 48 + k + lrow) * XSTRIDE + 16 * lhalf];

        // ---- 1-deep software-pipelined K loop: prefetch ci0+32 before WMMAs ----
        Frags cur, nxt;
        load_frags(cur, Ap0, Ap1, Bp0, Bp1, Bp2, Bp3, 0, lhalf);
#pragma unroll
        for (int ci0 = 0; ci0 < C_IN; ci0 += 32) {
            if (ci0 + 32 < C_IN)
                load_frags(nxt, Ap0, Ap1, Bp0, Bp1, Bp2, Bp3, ci0 + 32, lhalf);

            acc[0][0] = __builtin_amdgcn_wmma_f32_16x16x32_bf16(
                false, cur.a0.v, false, cur.b0.v, (short)0, acc[0][0], false, false);
            acc[1][0] = __builtin_amdgcn_wmma_f32_16x16x32_bf16(
                false, cur.a1.v, false, cur.b0.v, (short)0, acc[1][0], false, false);
            acc[0][1] = __builtin_amdgcn_wmma_f32_16x16x32_bf16(
                false, cur.a0.v, false, cur.b1.v, (short)0, acc[0][1], false, false);
            acc[1][1] = __builtin_amdgcn_wmma_f32_16x16x32_bf16(
                false, cur.a1.v, false, cur.b1.v, (short)0, acc[1][1], false, false);
            acc[0][2] = __builtin_amdgcn_wmma_f32_16x16x32_bf16(
                false, cur.a0.v, false, cur.b2.v, (short)0, acc[0][2], false, false);
            acc[1][2] = __builtin_amdgcn_wmma_f32_16x16x32_bf16(
                false, cur.a1.v, false, cur.b2.v, (short)0, acc[1][2], false, false);
            acc[0][3] = __builtin_amdgcn_wmma_f32_16x16x32_bf16(
                false, cur.a0.v, false, cur.b3.v, (short)0, acc[0][3], false, false);
            acc[1][3] = __builtin_amdgcn_wmma_f32_16x16x32_bf16(
                false, cur.a1.v, false, cur.b3.v, (short)0, acc[1][3], false, false);

            cur = nxt;   // register rename under full unroll; dead on last iteration
        }
    }

    // ---- epilogue: D layout VGPR r -> co = base + r (+8*lhalf), N col = lrow ----
#pragma unroll
    for (int i = 0; i < 2; ++i) {
        const int coBase = co0 + wco * 32 + i * 16 + 8 * lhalf;
        const float4 blo = *(const float4*)&bias[coBase];
        const float4 bhi = *(const float4*)&bias[coBase + 4];
#pragma unroll
        for (int j = 0; j < 4; ++j) {
            int t = t0 + wt * 64 + j * 16 + lrow;
            float* o = out + ((size_t)b * LEN + t) * C_OUT + coBase;
            v8f a = acc[i][j];
            float4 lo = make_float4(a[0] + blo.x, a[1] + blo.y,
                                    a[2] + blo.z, a[3] + blo.w);
            float4 hi = make_float4(a[4] + bhi.x, a[5] + bhi.y,
                                    a[6] + bhi.z, a[7] + bhi.w);
            *(float4*)o       = lo;
            *(float4*)(o + 4) = hi;
        }
    }
}

extern "C" void kernel_launch(void* const* d_in, const int* in_sizes, int n_in,
                              void* d_out, int out_size, void* d_ws, size_t ws_size,
                              hipStream_t stream) {
    const float* x    = (const float*)d_in[0];
    const float* w    = (const float*)d_in[1];
    const float* bias = (const float*)d_in[2];
    float* out        = (float*)d_out;

    unsigned short* w2 = (unsigned short*)d_ws;  // needs KTAPS*C_OUT*C_IN*2 = 2.36 MB

    const int wtotal = KTAPS * C_OUT * C_IN;
    repack_weights<<<(wtotal + 255) / 256, 256, 0, stream>>>(w, w2);

    dim3 grid(LEN / TILE_T, C_OUT / TILE_CO, BATCH);  // (32, 4, 32)
    conv1d_wmma_bf16<<<grid, 256, 0, stream>>>(x, w2, bias, out);
}
